// SelfAttentionBlock_31370441130170
// MI455X (gfx1250) — matrix operand order repaired and checked by reading, not compiled, not run on hardware
//
#include <hip/hip_runtime.h>
#include <hip/hip_bf16.h>

typedef __attribute__((ext_vector_type(16))) __bf16 v16bf;
typedef __attribute__((ext_vector_type(8)))  __bf16 v8bf;
typedef __attribute__((ext_vector_type(4)))  __bf16 v4bf;
typedef __attribute__((ext_vector_type(8)))  float  v8f;
typedef int v4i __attribute__((vector_size(16)));

#define BB 8
#define TT 2048
#define DD 512

#if defined(__gfx1250__) && __has_builtin(__builtin_amdgcn_global_load_async_to_lds_b128)
#define USE_ASYNC_LDS 1
#else
#define USE_ASYNC_LDS 0
#endif

// ---------------- WMMA helpers (CDNA5 bf16 16x16x32, fp32 accum) ----------------

__device__ __forceinline__ v8f wmma_bf16(v16bf a, v16bf b, v8f c) {
  // (neg_a, A, neg_b, B, c_mod, C, reuse_a, reuse_b)
  return __builtin_amdgcn_wmma_f32_16x16x32_bf16(false, a, false, b, (short)0, c,
                                                 false, false);
}

// A operand: 16x32 (MxK) bf16 from a row-major [16][ktot] strip (LDS or global).
// ISA layout: lanes 0-15 hold row M=lane with K = {0..7, 16..23};
//             lanes 16-31 hold row M=lane-16 with K = {8..15, 24..31}.
// => two contiguous 8-element (16B) loads per lane.
__device__ __forceinline__ v16bf load_a(const __bf16* base, int ktot, int k0,
                                        int lane) {
  int m  = lane & 15;
  int kb = (lane >> 4) << 3;          // 0 or 8
  const __bf16* p = base + (size_t)m * ktot + k0 + kb;
  v16bf a;
  ((v8bf*)&a)[0] = *(const v8bf*)p;
  ((v8bf*)&a)[1] = *(const v8bf*)(p + 16);
  return a;
}

// B operand: 32x16 (KxN) bf16. Source is stored "transposed" (row n = output
// column n), so each lane reads two 8-element (16B) K runs:
// lanes 0-15: column N=lane, K={0..7,16..23} ; lanes 16-31: column N=lane-16, K={8..15,24..31}.
__device__ __forceinline__ v16bf load_b(const __bf16* __restrict__ baseT, int ld,
                                        int col0, int k0, int lane) {
  int n  = lane & 15;
  int kb = (lane >> 4) << 3;          // 0 or 8
  const __bf16* p = baseT + (size_t)(col0 + n) * ld + k0 + kb;
  v16bf b;
  ((v8bf*)&b)[0] = *(const v8bf*)p;
  ((v8bf*)&b)[1] = *(const v8bf*)(p + 16);
  return b;
}

// Cooperative copy of the block-shared A strip (16 rows x ktot bf16, global row
// pitch `apitch` elements) into LDS, using CDNA5 async global->LDS DMA when the
// toolchain exposes it (ASYNCcnt path), else plain load/store.
__device__ __forceinline__ void stage_a_strip(__bf16* ldsA, const __bf16* g,
                                              size_t apitch, int row0, int ktot,
                                              int tid) {
  int chunksPerRow = ktot >> 3;       // 16B chunks per row
  int total = 16 * chunksPerRow;
  for (int id = tid; id < total; id += 256) {
    int r = id / chunksPerRow;
    int c = id - r * chunksPerRow;
    const __bf16* gp = g + (size_t)(row0 + r) * apitch + c * 8;
    __bf16* lp = ldsA + r * ktot + c * 8;
#if USE_ASYNC_LDS
    __builtin_amdgcn_global_load_async_to_lds_b128(
        (__attribute__((address_space(1))) v4i*)gp,
        (__attribute__((address_space(3))) v4i*)lp, 0, 0);
#else
    *(v8bf*)lp = *(const v8bf*)gp;
#endif
  }
}

__device__ __forceinline__ void wait_async_then_barrier() {
#if USE_ASYNC_LDS
#if __has_builtin(__builtin_amdgcn_s_wait_asynccnt)
  __builtin_amdgcn_s_wait_asynccnt(0);
#else
  asm volatile("s_wait_asynccnt 0x0" ::: "memory");
#endif
#endif
  __syncthreads();
}

// 16 x 64 GEMM strip: acc[0..3] += A_lds[16, 0:ktot] * Bt[col0+{0,16,32,48}, :]^T
// A comes from the LDS strip (ds_load_b128), B tiles stream from global.
__device__ __forceinline__ void gemm_strip(const __bf16* Alds, const __bf16* __restrict__ Bt,
                                           int ldb, int col0, int ktot, int lane,
                                           v8f acc[4]) {
  v16bf ac  = load_a(Alds, ktot, 0, lane);
  v16bf bc0 = load_b(Bt, ldb, col0,      0, lane);
  v16bf bc1 = load_b(Bt, ldb, col0 + 16, 0, lane);
  v16bf bc2 = load_b(Bt, ldb, col0 + 32, 0, lane);
  v16bf bc3 = load_b(Bt, ldb, col0 + 48, 0, lane);
#pragma unroll 2
  for (int kk = 32; kk < ktot; kk += 32) {
    v16bf an  = load_a(Alds, ktot, kk, lane);
    v16bf bn0 = load_b(Bt, ldb, col0,      kk, lane);
    v16bf bn1 = load_b(Bt, ldb, col0 + 16, kk, lane);
    v16bf bn2 = load_b(Bt, ldb, col0 + 32, kk, lane);
    v16bf bn3 = load_b(Bt, ldb, col0 + 48, kk, lane);
    acc[0] = wmma_bf16(ac, bc0, acc[0]);
    acc[1] = wmma_bf16(ac, bc1, acc[1]);
    acc[2] = wmma_bf16(ac, bc2, acc[2]);
    acc[3] = wmma_bf16(ac, bc3, acc[3]);
    ac = an; bc0 = bn0; bc1 = bn1; bc2 = bn2; bc3 = bn3;
  }
  acc[0] = wmma_bf16(ac, bc0, acc[0]);
  acc[1] = wmma_bf16(ac, bc1, acc[1]);
  acc[2] = wmma_bf16(ac, bc2, acc[2]);
  acc[3] = wmma_bf16(ac, bc3, acc[3]);
}

// ---------------- Prep kernels ----------------

__global__ __launch_bounds__(256)
void convert_x_kernel(const float* __restrict__ x, __bf16* __restrict__ xb) {
  int i = (blockIdx.x * 256 + threadIdx.x) * 4;
  float4 v = *(const float4*)(x + i);
  v4bf o;
  o[0] = (__bf16)v.x; o[1] = (__bf16)v.y; o[2] = (__bf16)v.z; o[3] = (__bf16)v.w;
  *(v4bf*)(xb + i) = o;
}

// Wt[w][n][k] = W_w[k][n]  (bf16, row n = column n of W)
__global__ __launch_bounds__(256)
void prep_w_kernel(const float* __restrict__ Wq, const float* __restrict__ Wk,
                   const float* __restrict__ Wv, __bf16* __restrict__ Wt) {
  int w = blockIdx.z;
  const float* W = (w == 0) ? Wq : (w == 1) ? Wk : Wv;
  int n = blockIdx.y;
  int k = blockIdx.x * 256 + threadIdx.x;
  Wt[((size_t)w * DD + n) * DD + k] = (__bf16)W[(size_t)k * DD + n];
}

// ---------------- QKV GEMM: [B*T,512] x [512,512] -> Q,K (bf16 row-major), V (bf16 transposed) ----------------

__global__ __launch_bounds__(256)
void qkv_kernel(const __bf16* __restrict__ xb, const __bf16* __restrict__ Wt,
                const float* __restrict__ bq, const float* __restrict__ bk,
                const float* __restrict__ bv,
                __bf16* __restrict__ Qb, __bf16* __restrict__ Kb,
                __bf16* __restrict__ Vt) {
  int lane = threadIdx.x & 31;
  int wave = threadIdx.x >> 5;
  int row0 = blockIdx.x * 16;           // global row in [0, B*T)
  int w    = blockIdx.y;                // 0=Q 1=K 2=V
  int col0 = wave * 64;
  const __bf16* Wm  = Wt + (size_t)w * DD * DD;
  const float* bias = (w == 0) ? bq : (w == 1) ? bk : bv;

  __shared__ __align__(16) __bf16 ldsA[16 * DD];       // 16KB A strip
  stage_a_strip(ldsA, xb, DD, row0, DD, threadIdx.x);
  wait_async_then_barrier();

  v8f acc[4] = {};
  gemm_strip(ldsA, Wm, DD, col0, DD, lane, acc);

  // C/D layout: lane holds column n = lane&15; VGPR r holds row m = r + 8*(lane>>4)
  int n = lane & 15;
  int mbase = (lane >> 4) * 8;
#pragma unroll
  for (int ti = 0; ti < 4; ++ti) {
    int col = col0 + ti * 16 + n;
    float bia = bias[col];
    if (w == 2) {
      // V transposed: Vt[b][col][t]; 16-row tile never crosses a batch boundary
      int bidx = row0 >> 11;
      int trow = (row0 & (TT - 1)) + mbase;
      v8bf pk;
#pragma unroll
      for (int rr = 0; rr < 8; ++rr) pk[rr] = (__bf16)(acc[ti][rr] + bia);
      *(v8bf*)(Vt + ((size_t)bidx * DD + col) * TT + trow) = pk;
    } else {
      __bf16* Out = (w == 0) ? Qb : Kb;
#pragma unroll
      for (int rr = 0; rr < 8; ++rr)
        Out[(size_t)(row0 + mbase + rr) * DD + col] = (__bf16)(acc[ti][rr] + bia);
    }
  }
}

// ---------------- Scores: S[b,t,s] = (Q . K^T) / sqrt(D)  (fp32) ----------------

__global__ __launch_bounds__(256)
void scores_kernel(const __bf16* __restrict__ Qb, const __bf16* __restrict__ Kb,
                   float* __restrict__ S) {
  int lane = threadIdx.x & 31;
  int wave = threadIdx.x >> 5;
  int b  = blockIdx.z;
  int t0 = blockIdx.x * 16;
  int s0 = blockIdx.y * 512 + wave * 64;
  const __bf16* Q = Qb + (size_t)b * TT * DD;
  const __bf16* K = Kb + (size_t)b * TT * DD;

  __shared__ __align__(16) __bf16 ldsA[16 * DD];       // 16KB Q strip
  stage_a_strip(ldsA, Q, DD, t0, DD, threadIdx.x);
  wait_async_then_barrier();

  // B = K^T: column s of B is row s of K -> row-major K works directly
  v8f acc[4] = {};
  gemm_strip(ldsA, K, DD, s0, DD, lane, acc);

  const float scale = 0.04419417382415922f;  // 1/sqrt(512)
  int n = lane & 15;
  int mbase = (lane >> 4) * 8;
  float* Sb = S + (size_t)b * TT * TT;
#pragma unroll
  for (int ti = 0; ti < 4; ++ti)
#pragma unroll
    for (int rr = 0; rr < 8; ++rr)
      Sb[(size_t)(t0 + mbase + rr) * TT + s0 + ti * 16 + n] = acc[ti][rr] * scale;
}

// ---------------- Softmax: fp32 row -> bf16 probs overlaid in-place ----------------

__global__ __launch_bounds__(256)
void softmax_kernel(float* __restrict__ S) {
  size_t gr = blockIdx.x;               // global row in [0, B*T)
  float* row = S + gr * TT;
  __bf16* prow = (__bf16*)row;          // overlay: first 4KB of the 8KB fp32 row
  int tid = threadIdx.x;

  float v[8];
  float mx = -3.4e38f;
#pragma unroll
  for (int i = 0; i < 8; ++i) { v[i] = row[tid + i * 256]; mx = fmaxf(mx, v[i]); }

  __shared__ float red[256];
  red[tid] = mx; __syncthreads();
  for (int s = 128; s > 0; s >>= 1) {
    if (tid < s) red[tid] = fmaxf(red[tid], red[tid + s]);
    __syncthreads();
  }
  mx = red[0]; __syncthreads();

  float sum = 0.f;
#pragma unroll
  for (int i = 0; i < 8; ++i) { v[i] = __expf(v[i] - mx); sum += v[i]; }
  red[tid] = sum; __syncthreads();
  for (int s = 128; s > 0; s >>= 1) {
    if (tid < s) red[tid] += red[tid + s];
    __syncthreads();
  }
  float inv = 1.0f / red[0];
  __syncthreads();
#pragma unroll
  for (int i = 0; i < 8; ++i) prow[tid + i * 256] = (__bf16)(v[i] * inv);
}

// ---------------- Context = P.V, + residual + LayerNorm (fused) ----------------

__global__ __launch_bounds__(256)
void context_ln_kernel(const float* __restrict__ S, const __bf16* __restrict__ Vt,
                       const float* __restrict__ x, const float* __restrict__ gamma,
                       const float* __restrict__ beta, float* __restrict__ out) {
  int lane = threadIdx.x & 31;
  int wave = threadIdx.x >> 5;
  int b  = blockIdx.y;
  int t0 = blockIdx.x * 16;
  int col0 = wave * 64;
  // P rows are bf16, overlaid at the start of each fp32 score row:
  // row pitch = TT*4 bytes = TT*2 bf16 elements.
  const __bf16* P = (const __bf16*)(S + (size_t)b * TT * TT);
  const __bf16* Vb = Vt + (size_t)b * DD * TT;

  // 64KB shared buffer: P strip (16 x 2048 bf16) during the GEMM, then reused
  // as the 32KB context/LN tile + 2KB reduction arrays.
  __shared__ __align__(16) char smem[16 * TT * 2];
  __bf16* ldsA = (__bf16*)smem;
  stage_a_strip(ldsA, P, (size_t)TT * 2, t0, TT, threadIdx.x);
  wait_async_then_barrier();

  v8f acc[4] = {};
  gemm_strip(ldsA, Vb, TT, col0, TT, lane, acc);
  __syncthreads();   // all ds reads of the P strip done before smem is reused

  float* tile = (float*)smem;                 // 16*512 fp32 = 32KB
  float* rs1  = (float*)(smem + 16 * DD * 4); // 1KB
  float* rs2  = rs1 + 256;                    // 1KB

  int n = lane & 15;
  int mbase = (lane >> 4) * 8;
  const float* xr = x + ((size_t)b * TT + t0) * DD;
#pragma unroll
  for (int ti = 0; ti < 4; ++ti) {
    int col = col0 + ti * 16 + n;
#pragma unroll
    for (int rr = 0; rr < 8; ++rr) {
      int m = mbase + rr;
      tile[m * DD + col] = acc[ti][rr] + xr[(size_t)m * DD + col];
    }
  }
  __syncthreads();

  // LayerNorm: 16 threads per row, each sums 32 elements
  int r = threadIdx.x >> 4;
  int c = threadIdx.x & 15;
  float s1 = 0.f, s2 = 0.f;
#pragma unroll
  for (int i = 0; i < 32; ++i) {
    float u = tile[r * DD + c + i * 16];
    s1 += u; s2 += u * u;
  }
  rs1[threadIdx.x] = s1; rs2[threadIdx.x] = s2;
  __syncthreads();
  for (int s = 8; s > 0; s >>= 1) {
    if (c < s) {
      rs1[threadIdx.x] += rs1[threadIdx.x + s];
      rs2[threadIdx.x] += rs2[threadIdx.x + s];
    }
    __syncthreads();
  }
  float mean = rs1[r * 16] * (1.0f / DD);
  float var  = rs2[r * 16] * (1.0f / DD) - mean * mean;
  float rstd = rsqrtf(var + 1e-3f);

  float* ob = out + ((size_t)b * TT + t0) * DD;
#pragma unroll
  for (int i = 0; i < 32; ++i) {
    int col = c + i * 16;
    ob[(size_t)r * DD + col] =
        (tile[r * DD + col] - mean) * rstd * gamma[col] + beta[col];
  }
}

// ---------------- launch ----------------

extern "C" void kernel_launch(void* const* d_in, const int* in_sizes, int n_in,
                              void* d_out, int out_size, void* d_ws, size_t ws_size,
                              hipStream_t stream) {
  const float* x     = (const float*)d_in[0];
  const float* Wq    = (const float*)d_in[1];
  const float* bq    = (const float*)d_in[2];
  const float* Wk    = (const float*)d_in[3];
  const float* bk    = (const float*)d_in[4];
  const float* Wv    = (const float*)d_in[5];
  const float* bv    = (const float*)d_in[6];
  const float* gamma = (const float*)d_in[7];
  const float* beta  = (const float*)d_in[8];
  float* out = (float*)d_out;

  char* ws = (char*)d_ws;
  size_t off = 0;
  __bf16* xb = (__bf16*)(ws + off); off += (size_t)BB * TT * DD * 2;
  __bf16* Wt = (__bf16*)(ws + off); off += (size_t)3 * DD * DD * 2;
  __bf16* Qb = (__bf16*)(ws + off); off += (size_t)BB * TT * DD * 2;
  __bf16* Kb = (__bf16*)(ws + off); off += (size_t)BB * TT * DD * 2;
  __bf16* Vt = (__bf16*)(ws + off); off += (size_t)BB * TT * DD * 2;
  float*  S  = (float*)(ws + off);  off += (size_t)BB * TT * TT * 4;

  int nx = BB * TT * DD;
  convert_x_kernel<<<nx / 1024, 256, 0, stream>>>(x, xb);
  prep_w_kernel<<<dim3(DD / 256, DD, 3), 256, 0, stream>>>(Wq, Wk, Wv, Wt);
  qkv_kernel<<<dim3(BB * TT / 16, 3), 256, 0, stream>>>(xb, Wt, bq, bk, bv, Qb, Kb, Vt);
  scores_kernel<<<dim3(TT / 16, TT / 512, BB), 256, 0, stream>>>(Qb, Kb, S);
  softmax_kernel<<<BB * TT, 256, 0, stream>>>(S);
  context_ln_kernel<<<dim3(TT / 16, BB), 256, 0, stream>>>(S, Vt, x, gamma, beta, out);
}
